// LongShortTermTransformer_43722767073722
// MI455X (gfx1250) — compile-verified, matrix operand order checked by reading, not run on hardware
//
#include <hip/hip_runtime.h>
#include <hip/hip_bf16.h>

typedef __attribute__((ext_vector_type(16))) __bf16 v16bf;
typedef __attribute__((ext_vector_type(8)))  __bf16 v8bf;
typedef __attribute__((ext_vector_type(8)))  float  v8f;

constexpr int kHW = 1024;   // 32x32 spatial
constexpr int kB  = 4;
constexpr int kC  = 256;
constexpr int kFF = 1024;
constexpr int kH  = 8;
constexpr int kD  = 32;     // head dim
constexpr int kNL = 2;
constexpr int kM  = kHW * kB;   // 4096 tokens, token t = hw*B + b

// ---------------------------------------------------------------------------
// WMMA helpers (CDNA5 gfx1250, wave32, v_wmma_f32_16x16x32_bf16)
// ---------------------------------------------------------------------------
__device__ __forceinline__ v8f wmma_bf16(v16bf a, v16bf b, v8f c) {
  return __builtin_amdgcn_wmma_f32_16x16x32_bf16(false, a, false, b, (short)0, c,
                                                 false, false);
}

// Load a 16x32 bf16 A-fragment from row-major memory (row stride ld elements).
// Per ISA 7.12.2: lane l holds row (l&15); VGPRs 0-3 hold K = (l>>4)*8 + 0..7,
// VGPRs 4-7 hold K = 16 + (l>>4)*8 + 0..7.  A B-fragment of a KxN matrix
// stored as its [N,K] transpose uses the identical addressing (n = l&15).
__device__ __forceinline__ v16bf load_frag(const __bf16* __restrict__ p0, int ld) {
  int lane = threadIdx.x & 31;
  int r = lane & 15;
  int h = (lane >> 4) << 3;
  const __bf16* p = p0 + (long)r * ld + h;
  v8bf lo = *(const v8bf*)(p);
  v8bf hi = *(const v8bf*)(p + 16);
  v16bf f;
#pragma unroll
  for (int i = 0; i < 8; ++i) { f[i] = lo[i]; f[i + 8] = hi[i]; }
  return f;
}

// Same fragment pattern from an LDS tile with row stride 32 bf16.
__device__ __forceinline__ v16bf load_frag_lds(const __bf16* pb, int lane) {
  int r = lane & 15;
  int h = (lane >> 4) << 3;
  v16bf f;
#pragma unroll
  for (int i = 0; i < 8; ++i) {
    f[i]     = pb[r * 32 + h + i];
    f[i + 8] = pb[r * 32 + 16 + h + i];
  }
  return f;
}

// 32-bit LDS offset of a generic pointer into a __shared__ object (generic LDS
// pointers carry the aperture in the high bits; addr[31:0] is the LDS offset).
__device__ __forceinline__ unsigned lds_off(const void* p) {
  return (unsigned)(unsigned long long)p;
}

// CDNA5 async copy: 16B global -> LDS, tracked by ASYNCcnt (no VGPR staging).
__device__ __forceinline__ void async_copy_b128(unsigned dst_lds, const void* src) {
  asm volatile("global_load_async_to_lds_b128 %0, %1, off"
               :
               : "v"(dst_lds), "v"(src)
               : "memory");
}

// ---------------------------------------------------------------------------
// GEMM:  out[M,N] = A[M,K](bf16) * BT[N,K]^T(bf16) + bias + resid
// 256 threads = 8 waves; wave (wm,wn) computes 16x64; block tile 64x128.
// A/B tiles double-buffered in LDS via global_load_async_to_lds_b128;
// next tile's copies overlap current tile's WMMAs (s_wait_asynccnt).
// ---------------------------------------------------------------------------
__global__ __launch_bounds__(256) void gemm_bias_kernel(
    const __bf16* __restrict__ A, const __bf16* __restrict__ BT,
    const float* __restrict__ bias, const float* __restrict__ resid,
    float* __restrict__ outF, __bf16* __restrict__ outB,
    int M, int N, int K)
{
  __shared__ __bf16 lA[2][64 * 32];    // [buf][m*32 + k]   4 KB each
  __shared__ __bf16 lB[2][128 * 32];   // [buf][n*32 + k]   8 KB each
  int tid = threadIdx.x;
  int wave = tid >> 5;
  int lane = tid & 31;
  int wm = wave & 3, wn = wave >> 2;
  int m0 = blockIdx.x * 64;
  int n0 = blockIdx.y * 128;

  // Cooperative staging roles: 16B chunks, 8 bf16 each.
  int srow = tid >> 2;             // 0..63
  int schk = (tid & 3) * 8;        // 0,8,16,24 (bf16 elems)

  auto stage = [&](int buf, int k0) {
    async_copy_b128(lds_off(&lA[buf][srow * 32 + schk]),
                    A + (long)(m0 + srow) * K + k0 + schk);
    async_copy_b128(lds_off(&lB[buf][srow * 32 + schk]),
                    BT + (long)(n0 + srow) * K + k0 + schk);
    async_copy_b128(lds_off(&lB[buf][(srow + 64) * 32 + schk]),
                    BT + (long)(n0 + srow + 64) * K + k0 + schk);
  };

  v8f acc[4] = {};
  const int nk = K >> 5;
  stage(0, 0);
  for (int ki = 0; ki < nk; ++ki) {
    int buf = ki & 1;
    if (ki + 1 < nk) {
      stage(buf ^ 1, (ki + 1) << 5);
      asm volatile("s_wait_asynccnt 0x3" ::: "memory");  // current buf's 3 done
    } else {
      asm volatile("s_wait_asynccnt 0x0" ::: "memory");
    }
    __syncthreads();
    v16bf a = load_frag_lds(&lA[buf][(wm * 16) * 32], lane);
#pragma unroll
    for (int t = 0; t < 4; ++t) {
      v16bf b = load_frag_lds(&lB[buf][(wn * 64 + t * 16) * 32], lane);
      acc[t] = wmma_bf16(a, b, acc[t]);
    }
    __syncthreads();  // all waves done reading buf before it is re-staged
  }

  // C layout: VGPR j, lanes 0-15 -> M=m0+j, lanes 16-31 -> M=m0+8+j; N=n0+(lane&15)
  int cn = lane & 15;
  int rbase = m0 + wm * 16 + ((lane >> 4) << 3);
#pragma unroll
  for (int t = 0; t < 4; ++t) {
    int n = n0 + wn * 64 + t * 16 + cn;
    float bv = bias ? bias[n] : 0.f;
#pragma unroll
    for (int j = 0; j < 8; ++j) {
      long idx = (long)(rbase + j) * N + n;
      float v = acc[t][j] + bv;
      if (resid) v += resid[idx];
      if (outF) outF[idx] = v;
      if (outB) outB[idx] = (__bf16)v;
    }
  }
}

// ---------------------------------------------------------------------------
// Fused flash attention.  Q,K: bf16 token-major [L, B, C]; V: bf16 [B,H,d,Lk].
// One wave per 16-row query tile; d = 32 => one A-fragment per Q tile.
// ---------------------------------------------------------------------------
__global__ __launch_bounds__(128) void attn_kernel(
    const __bf16* __restrict__ Q, const __bf16* __restrict__ Kb,
    const __bf16* __restrict__ Vt, __bf16* __restrict__ O,
    int Lk, float scale)
{
  __shared__ __bf16 pbuf[4][16 * 32];
  int wave = threadIdx.x >> 5;
  int lane = threadIdx.x & 31;
  int b = blockIdx.y, h = blockIdx.z;
  int q0 = blockIdx.x * 64 + wave * 16;

  const int ldTok = kB * kC;
  v16bf aq = load_frag(Q + ((long)q0 * kB + b) * kC + h * kD, ldTok);

  v8f acc0 = {}, acc1 = {};
  float mrun[8], lrun[8];
#pragma unroll
  for (int j = 0; j < 8; ++j) { mrun[j] = -1e30f; lrun[j] = 0.f; }

  const float c144 = 1.44269504f;
  __bf16* pb = pbuf[wave];
  const __bf16* vbase = Vt + ((long)(b * kH + h) * kD) * Lk;
  int row_off = (lane >> 4) << 3;
  int colp = lane & 15;

  for (int kk = 0; kk < Lk; kk += 32) {
    if (kk + 32 < Lk) {  // pull next K/V chunk toward L2 (global_prefetch_b8)
      __builtin_prefetch(Kb + ((long)(kk + 32) * kB + b) * kC + h * kD, 0, 1);
      __builtin_prefetch(vbase + kk + 32, 0, 1);
    }
    v16bf bk0 = load_frag(Kb + ((long)kk * kB + b) * kC + h * kD, ldTok);
    v16bf bk1 = load_frag(Kb + ((long)(kk + 16) * kB + b) * kC + h * kD, ldTok);
    v8f z = {};
    v8f s0 = wmma_bf16(aq, bk0, z);
    v8f s1 = wmma_bf16(aq, bk1, z);

    float p0[8], p1[8];
#pragma unroll
    for (int j = 0; j < 8; ++j) {
      float x0 = s0[j] * scale, x1 = s1[j] * scale;
      float mx = fmaxf(x0, x1);
#pragma unroll
      for (int m = 1; m < 16; m <<= 1) mx = fmaxf(mx, __shfl_xor(mx, m));
      float mnew = fmaxf(mrun[j], mx);
      float corr = exp2f((mrun[j] - mnew) * c144);
      p0[j] = exp2f((x0 - mnew) * c144);
      p1[j] = exp2f((x1 - mnew) * c144);
      float sum = p0[j] + p1[j];
#pragma unroll
      for (int m = 1; m < 16; m <<= 1) sum += __shfl_xor(sum, m);
      lrun[j] = lrun[j] * corr + sum;
      mrun[j] = mnew;
      acc0[j] *= corr;
      acc1[j] *= corr;
    }
    // Repack P (C layout) -> A-fragment layout through per-wave LDS tile.
#pragma unroll
    for (int j = 0; j < 8; ++j) {
      pb[(j + row_off) * 32 + colp]      = (__bf16)p0[j];
      pb[(j + row_off) * 32 + 16 + colp] = (__bf16)p1[j];
    }
    asm volatile("s_wait_dscnt 0" ::: "memory");
    v16bf pf = load_frag_lds(pb, lane);

    v16bf bv0 = load_frag(vbase + kk, Lk);                 // d = 0..15
    v16bf bv1 = load_frag(vbase + (long)16 * Lk + kk, Lk); // d = 16..31
    acc0 = wmma_bf16(pf, bv0, acc0);
    acc1 = wmma_bf16(pf, bv1, acc1);
  }

#pragma unroll
  for (int j = 0; j < 8; ++j) {
    float inv = 1.f / lrun[j];
    long obase = ((long)(q0 + j + row_off) * kB + b) * kC + h * kD;
    O[obase + colp]      = (__bf16)(acc0[j] * inv);
    O[obase + 16 + colp] = (__bf16)(acc1[j] * inv);
  }
}

// ---------------------------------------------------------------------------
// LayerNorm over C=256 (one wave per token, 8 waves per block).
// out = LN(X + addPre);  outB2 = bf16(out + addPost).
// ---------------------------------------------------------------------------
__global__ __launch_bounds__(256) void ln_kernel(
    const float* __restrict__ X, const float* __restrict__ addPre,
    const float* __restrict__ g, const float* __restrict__ be,
    float* __restrict__ outF, __bf16* __restrict__ outB,
    const float* __restrict__ addPost, __bf16* __restrict__ outB2, int M)
{
  int t = blockIdx.x * 8 + (threadIdx.x >> 5);
  int lane = threadIdx.x & 31;
  if (t >= M) return;
  long base = (long)t * kC + lane * 8;
  float v[8];
  float s = 0.f;
#pragma unroll
  for (int i = 0; i < 8; ++i) {
    v[i] = X[base + i] + (addPre ? addPre[base + i] : 0.f);
    s += v[i];
  }
#pragma unroll
  for (int m = 1; m < 32; m <<= 1) s += __shfl_xor(s, m);
  float mean = s * (1.f / kC);
  float s2 = 0.f;
#pragma unroll
  for (int i = 0; i < 8; ++i) { float d = v[i] - mean; s2 += d * d; }
#pragma unroll
  for (int m = 1; m < 32; m <<= 1) s2 += __shfl_xor(s2, m);
  float rstd = rsqrtf(s2 * (1.f / kC) + 1e-5f);
#pragma unroll
  for (int i = 0; i < 8; ++i) {
    int c = lane * 8 + i;
    float o = (v[i] - mean) * rstd * g[c] + be[c];
    if (outF) outF[base + i] = o;
    if (outB) outB[base + i] = (__bf16)o;
    if (outB2) outB2[base + i] = (__bf16)(o + addPost[base + i]);
  }
}

// ---------------------------------------------------------------------------
// Transposes (32x32 LDS tiles)
// ---------------------------------------------------------------------------
// V [L,B,C](TIn) -> Vt [B,H,d=32,L] bf16
template <typename TIn>
__global__ __launch_bounds__(256) void transpose_v_kernel(
    const TIn* __restrict__ in, __bf16* __restrict__ out, int L)
{
  __shared__ float tile[32][33];
  int l0 = blockIdx.x * 32;
  int bh = blockIdx.y;
  int b = bh / kH, h = bh % kH;
  int tx = threadIdx.x, ty = threadIdx.y;
#pragma unroll
  for (int i = 0; i < 4; ++i) {
    int l = l0 + ty + i * 8;
    tile[ty + i * 8][tx] = (float)in[((long)l * kB + b) * kC + h * kD + tx];
  }
  __syncthreads();
#pragma unroll
  for (int i = 0; i < 4; ++i) {
    int d = ty + i * 8;
    out[((long)bh * kD + d) * L + l0 + tx] = (__bf16)tile[tx][d];
  }
}

// W [K,N] f32 -> WT [N,K] bf16
__global__ __launch_bounds__(256) void wt_kernel(
    const float* __restrict__ W, __bf16* __restrict__ WT, int K, int N)
{
  __shared__ float tile[32][33];
  int k0 = blockIdx.x * 32, n0 = blockIdx.y * 32;
  int tx = threadIdx.x, ty = threadIdx.y;
#pragma unroll
  for (int i = 0; i < 4; ++i)
    tile[ty + i * 8][tx] = W[(long)(k0 + ty + i * 8) * N + n0 + tx];
  __syncthreads();
#pragma unroll
  for (int i = 0; i < 4; ++i)
    WT[(long)(n0 + ty + i * 8) * K + k0 + tx] = (__bf16)tile[tx][ty + i * 8];
}

__global__ void cast_bf16_kernel(const float* __restrict__ in,
                                 __bf16* __restrict__ out, long n)
{
  for (long i = (long)blockIdx.x * blockDim.x + threadIdx.x; i < n;
       i += (long)gridDim.x * blockDim.x)
    out[i] = (__bf16)in[i];
}

// ---------------------------------------------------------------------------
// GroupNorm(32 groups) + exact GELU + depthwise 5x5 conv over 32x32 spatial
// ---------------------------------------------------------------------------
__global__ __launch_bounds__(256) void gn_stats_kernel(
    const float* __restrict__ F, float* __restrict__ stats)
{
  int b = blockIdx.x >> 5, g = blockIdx.x & 31;
  int lane = threadIdx.x & 31, wave = threadIdx.x >> 5;
  float s = 0.f, s2 = 0.f;
  for (int i = threadIdx.x; i < kHW * 32; i += 256) {
    int hw = i >> 5, cc = i & 31;
    float v = F[((long)hw * kB + b) * kFF + g * 32 + cc];
    s += v; s2 += v * v;
  }
#pragma unroll
  for (int m = 1; m < 32; m <<= 1) { s += __shfl_xor(s, m); s2 += __shfl_xor(s2, m); }
  __shared__ float rs[8], rq[8];
  if (lane == 0) { rs[wave] = s; rq[wave] = s2; }
  __syncthreads();
  if (threadIdx.x == 0) {
    float ts = 0.f, tq = 0.f;
    for (int i = 0; i < 8; ++i) { ts += rs[i]; tq += rq[i]; }
    float mean = ts / (float)(kHW * 32);
    float var = tq / (float)(kHW * 32) - mean * mean;
    stats[blockIdx.x * 2] = mean;
    stats[blockIdx.x * 2 + 1] = rsqrtf(var + 1e-5f);
  }
}

__global__ __launch_bounds__(256) void gn_gelu_kernel(
    float* __restrict__ F, const float* __restrict__ stats,
    const float* __restrict__ gg, const float* __restrict__ gb)
{
  long i = (long)blockIdx.x * 256 + threadIdx.x;  // over M*FF
  int ch = (int)(i & (kFF - 1));
  int m = (int)(i >> 10);
  int b = m & (kB - 1);
  int g = ch >> 5;
  float mean = stats[(b * 32 + g) * 2];
  float rstd = stats[(b * 32 + g) * 2 + 1];
  float y = (F[i] - mean) * rstd * gg[ch] + gb[ch];
  y = 0.5f * y * (1.f + erff(y * 0.70710678118f));
  F[i] = y;
}

__global__ __launch_bounds__(256) void dwconv_kernel(
    const float* __restrict__ F, const float* __restrict__ dw,
    __bf16* __restrict__ G)
{
  long i = (long)blockIdx.x * 256 + threadIdx.x;
  int ch = (int)(i & (kFF - 1));
  int m = (int)(i >> 10);
  int b = m & (kB - 1);
  int hw = m >> 2;
  int hh = hw >> 5, ww = hw & 31;
  float s = 0.f;
#pragma unroll
  for (int kh = 0; kh < 5; ++kh) {
    int h2 = hh + kh - 2;
    if (h2 < 0 || h2 > 31) continue;
#pragma unroll
    for (int kw = 0; kw < 5; ++kw) {
      int w2 = ww + kw - 2;
      if (w2 < 0 || w2 > 31) continue;
      s += F[(((long)(h2 * 32 + w2) * kB + b) << 10) + ch] * dw[ch * 25 + kh * 5 + kw];
    }
  }
  G[i] = (__bf16)s;
}

// ---------------------------------------------------------------------------
// Host orchestration
// ---------------------------------------------------------------------------
extern "C" void kernel_launch(void* const* d_in, const int* in_sizes, int n_in,
                              void* d_out, int out_size, void* d_ws, size_t ws_size,
                              hipStream_t stream)
{
  (void)out_size; (void)ws_size;
  // Canonical (insertion-order) per-layer leaves:
  //  0 ln1_g  1 ln1_b  2 sa_Wq  3 sa_bq  4 sa_Wk  5 sa_bk  6 sa_Wv  7 sa_bv
  //  8 sa_Wo  9 sa_bo 10 ln2_g 11 ln2_b 12 Wq    13 bq    14 Wqm   15 bqm
  // 16 ln4_g 17 ln4_b 18 lt_Wo 19 lt_bo 20 st_Wo 21 st_bo 22 ln3_g 23 ln3_b
  // 24 W1    25 b1    26 gn_g  27 gn_b  28 dw    29 W2    30 b2
  static const int leafSizes[31] = {
      256, 256, 65536, 256, 65536, 256, 65536, 256, 65536, 256, 256, 256,
      65536, 256, 65536, 256, 256, 256, 65536, 256, 65536, 256, 256, 256,
      262144, 1024, 1024, 1024, 25600, 262144, 256};
  static const int sortedToCanon[31] = {
      24, 29, 12, 14, 25, 30, 13, 15, 28, 27, 26, 1, 0, 11, 10, 23,
      22, 17, 16, 18, 19, 4, 8, 2, 6, 5, 9, 3, 7, 20, 21};

  const float *tgt = nullptr, *longK = nullptr, *longV = nullptr;
  const float *shortK = nullptr, *shortV = nullptr, *selfpos = nullptr;
  const float* L[kNL][31];
  const float *lnfg = nullptr, *lnfb = nullptr;

  const long bankElems = (long)4 * kHW * kB * kC;  // T*HW*B*C per layer

  if (n_in >= 70) {
    if (in_sizes[0] == (int)(kNL * bankElems)) {
      // full jax tree_flatten: sorted keys at every level
      longK = (const float*)d_in[0];
      longV = (const float*)d_in[1];
      for (int l = 0; l < kNL; ++l)
        for (int i = 0; i < 31; ++i)
          L[l][sortedToCanon[i]] = (const float*)d_in[2 + l * 31 + i];
      lnfb = (const float*)d_in[64];
      lnfg = (const float*)d_in[65];
      selfpos = (const float*)d_in[66];
      shortK = (const float*)d_in[67];
      shortV = (const float*)d_in[68];
      tgt = (const float*)d_in[69];
    } else {
      tgt = (const float*)d_in[0];
      longK = (const float*)d_in[1];
      longV = (const float*)d_in[2];
      shortK = (const float*)d_in[3];
      shortV = (const float*)d_in[4];
      selfpos = (const float*)d_in[5];
      if (in_sizes[6] == 256) {  // insertion-order params (ln1_g first)
        for (int l = 0; l < kNL; ++l)
          for (int i = 0; i < 31; ++i)
            L[l][i] = (const float*)d_in[6 + l * 31 + i];
        lnfg = (const float*)d_in[68];
        lnfb = (const float*)d_in[69];
      } else {  // sorted params (W1 first)
        for (int l = 0; l < kNL; ++l)
          for (int i = 0; i < 31; ++i)
            L[l][sortedToCanon[i]] = (const float*)d_in[6 + l * 31 + i];
        lnfb = (const float*)d_in[68];
        lnfg = (const float*)d_in[69];
      }
    }
  } else {
    // params concatenated into one blob, insertion order assumed
    tgt = (const float*)d_in[0];
    longK = (const float*)d_in[1];
    longV = (const float*)d_in[2];
    shortK = (const float*)d_in[3];
    shortV = (const float*)d_in[4];
    selfpos = (const float*)d_in[5];
    const float* p = (const float*)d_in[6];
    for (int l = 0; l < kNL; ++l)
      for (int i = 0; i < 31; ++i) { L[l][i] = p; p += leafSizes[i]; }
    lnfg = p; p += 256; lnfb = p;
  }

  // ---- workspace bump allocation ----
  size_t off = 0;
  char* wsb = (char*)d_ws;
  auto alloc = [&](size_t bytes) -> void* {
    void* p = wsb + off;
    off = (off + bytes + 255) & ~(size_t)255;
    return p;
  };

  float* X = (float*)alloc((size_t)kM * kC * 4);
  struct WTL { __bf16 *Wq, *Wk, *Wv, *Wo, *Wcq, *ltWo, *stWo, *W1t, *W2t; } wt[kNL];
  __bf16 *LKb[kNL], *LVt[kNL];
  for (int l = 0; l < kNL; ++l) {
    wt[l].Wq   = (__bf16*)alloc((size_t)kC * kC * 2);
    wt[l].Wk   = (__bf16*)alloc((size_t)kC * kC * 2);
    wt[l].Wv   = (__bf16*)alloc((size_t)kC * kC * 2);
    wt[l].Wo   = (__bf16*)alloc((size_t)kC * kC * 2);
    wt[l].Wcq  = (__bf16*)alloc((size_t)kC * kC * 2);
    wt[l].ltWo = (__bf16*)alloc((size_t)kC * kC * 2);
    wt[l].stWo = (__bf16*)alloc((size_t)kC * kC * 2);
    wt[l].W1t  = (__bf16*)alloc((size_t)kC * kFF * 2);
    wt[l].W2t  = (__bf16*)alloc((size_t)kFF * kC * 2);
    LKb[l] = (__bf16*)alloc((size_t)bankElems * 2);
    LVt[l] = (__bf16*)alloc((size_t)bankElems * 2);
  }
  __bf16* aQK = (__bf16*)alloc((size_t)kM * kC * 2);
  __bf16* aV  = (__bf16*)alloc((size_t)kM * kC * 2);
  __bf16* Qb  = (__bf16*)alloc((size_t)kM * kC * 2);
  __bf16* Kbf = (__bf16*)alloc((size_t)kM * kC * 2);
  __bf16* Vb  = (__bf16*)alloc((size_t)kM * kC * 2);
  __bf16* Vt  = (__bf16*)alloc((size_t)kM * kC * 2);
  __bf16* AO  = (__bf16*)alloc((size_t)kM * kC * 2);
  __bf16* Qcb = (__bf16*)alloc((size_t)kM * kC * 2);
  __bf16* SKb = (__bf16*)alloc((size_t)kM * kC * 2);
  __bf16* SVb = (__bf16*)alloc((size_t)kM * kC * 2);
  float* T2F  = (float*)alloc((size_t)kM * kC * 4);
  float* QcF  = (float*)alloc((size_t)kM * kC * 4);
  float* FF1  = (float*)alloc((size_t)kM * kFF * 4);
  __bf16* FFG = (__bf16*)alloc((size_t)kM * kFF * 2);
  float* STATS = (float*)alloc((size_t)kB * 32 * 2 * 4);

  // ---- prep: x = tgt, weights -> bf16 transposed, long banks -> bf16 ----
  hipMemcpyAsync(X, tgt, (size_t)kM * kC * 4, hipMemcpyDeviceToDevice, stream);
  dim3 t328(32, 8);
  for (int l = 0; l < kNL; ++l) {
    wt_kernel<<<dim3(kC / 32, kC / 32), t328, 0, stream>>>(L[l][2],  wt[l].Wq,  kC, kC);
    wt_kernel<<<dim3(kC / 32, kC / 32), t328, 0, stream>>>(L[l][4],  wt[l].Wk,  kC, kC);
    wt_kernel<<<dim3(kC / 32, kC / 32), t328, 0, stream>>>(L[l][6],  wt[l].Wv,  kC, kC);
    wt_kernel<<<dim3(kC / 32, kC / 32), t328, 0, stream>>>(L[l][8],  wt[l].Wo,  kC, kC);
    wt_kernel<<<dim3(kC / 32, kC / 32), t328, 0, stream>>>(L[l][12], wt[l].Wcq, kC, kC);
    wt_kernel<<<dim3(kC / 32, kC / 32), t328, 0, stream>>>(L[l][18], wt[l].ltWo, kC, kC);
    wt_kernel<<<dim3(kC / 32, kC / 32), t328, 0, stream>>>(L[l][20], wt[l].stWo, kC, kC);
    wt_kernel<<<dim3(kC / 32, kFF / 32), t328, 0, stream>>>(L[l][24], wt[l].W1t, kC, kFF);
    wt_kernel<<<dim3(kFF / 32, kC / 32), t328, 0, stream>>>(L[l][29], wt[l].W2t, kFF, kC);
    cast_bf16_kernel<<<2048, 256, 0, stream>>>(longK + (size_t)l * bankElems, LKb[l], bankElems);
    transpose_v_kernel<float><<<dim3(4 * kHW / 32, kB * kH), t328, 0, stream>>>(
        longV + (size_t)l * bankElems, LVt[l], 4 * kHW);
  }

  const float scale = 0.17677669529663687f;  // 1/sqrt(32)
  auto gemm = [&](const __bf16* A, const __bf16* BT, const float* bias,
                  const float* resid, float* oF, __bf16* oB, int N, int K) {
    gemm_bias_kernel<<<dim3(kM / 64, N / 128), 256, 0, stream>>>(
        A, BT, bias, resid, oF, oB, kM, N, K);
  };
  auto attn = [&](const __bf16* Qp, const __bf16* Kp, const __bf16* Vp,
                  __bf16* Op, int Lk) {
    attn_kernel<<<dim3(kHW / 64, kB, kH), 128, 0, stream>>>(Qp, Kp, Vp, Op, Lk, scale);
  };

  for (int l = 0; l < kNL; ++l) {
    const float* const* P = L[l];
    // --- self attention ---
    ln_kernel<<<kM / 8, 256, 0, stream>>>(X, nullptr, P[0], P[1], nullptr, aV,
                                          selfpos, aQK, kM);
    gemm(aQK, wt[l].Wq, P[3], nullptr, nullptr, Qb, kC, kC);
    gemm(aQK, wt[l].Wk, P[5], nullptr, nullptr, Kbf, kC, kC);
    gemm(aV,  wt[l].Wv, P[7], nullptr, nullptr, Vb, kC, kC);
    transpose_v_kernel<__bf16><<<dim3(kHW / 32, kB * kH), t328, 0, stream>>>(Vb, Vt, kHW);
    attn(Qb, Kbf, Vt, AO, kHW);
    gemm(AO, wt[l].Wo, P[9], X, X, nullptr, kC, kC);
    // --- long/short term cross attention ---
    ln_kernel<<<kM / 8, 256, 0, stream>>>(X, nullptr, P[10], P[11], T2F, aQK,
                                          nullptr, nullptr, kM);
    gemm(aQK, wt[l].Wcq, P[13], nullptr, QcF, Qcb, kC, kC);
    attn(Qcb, LKb[l], LVt[l], AO, 4 * kHW);
    gemm(AO, wt[l].ltWo, P[19], X, X, nullptr, kC, kC);
    ln_kernel<<<kM / 8, 256, 0, stream>>>(QcF, shortK + (size_t)l * kM * kC,
                                          P[16], P[17], nullptr, SKb, nullptr, nullptr, kM);
    ln_kernel<<<kM / 8, 256, 0, stream>>>(T2F, shortV + (size_t)l * kM * kC,
                                          P[16], P[17], nullptr, SVb, nullptr, nullptr, kM);
    transpose_v_kernel<__bf16><<<dim3(kHW / 32, kB * kH), t328, 0, stream>>>(SVb, Vt, kHW);
    attn(Qcb, SKb, Vt, AO, kHW);
    gemm(AO, wt[l].stWo, P[21], X, X, nullptr, kC, kC);
    // --- FFN with GN + GELU + depthwise conv ---
    ln_kernel<<<kM / 8, 256, 0, stream>>>(X, nullptr, P[22], P[23], nullptr, aQK,
                                          nullptr, nullptr, kM);
    gemm(aQK, wt[l].W1t, P[25], nullptr, FF1, nullptr, kFF, kC);
    gn_stats_kernel<<<kB * 32, 256, 0, stream>>>(FF1, STATS);
    gn_gelu_kernel<<<(kM * kFF) / 256, 256, 0, stream>>>(FF1, STATS, P[26], P[27]);
    dwconv_kernel<<<(kM * kFF) / 256, 256, 0, stream>>>(FF1, P[28], FFG);
    gemm(FFG, wt[l].W2t, P[30], X, X, nullptr, kC, kFF);
  }
  ln_kernel<<<kM / 8, 256, 0, stream>>>(X, nullptr, lnfg, lnfb, (float*)d_out,
                                        nullptr, nullptr, nullptr, kM);
}
